// GCN_31104153158278
// MI455X (gfx1250) — matrix operand order repaired and checked by reading, not compiled, not run on hardware
//
#include <hip/hip_runtime.h>
#include <hip/hip_bf16.h>

// GCN 2-layer forward for MI455X (gfx1250, wave32).
// GEMMs via v_wmma_f32_16x16x32_f16 (f16 in, f32 acc); graph aggregation via
// float4 gathers + global_atomic_add_f32. Memory-bound workload (~1.2 GB moved).
// GEMM1 keeps register pressure ~90 VGPRs (no spills) and relies on 8-wave
// occupancy + batched B-fragment loads for latency hiding.

typedef __attribute__((ext_vector_type(16))) _Float16 v16h;
typedef __attribute__((ext_vector_type(8)))  _Float16 v8h;
typedef __attribute__((ext_vector_type(8)))  float    v8f;
typedef __attribute__((ext_vector_type(4)))  float    v4f;

#define F_IN 512
#define HID  64
#define NCLS 16

// ---------------------------------------------------------------------------
// init: deg=1.0 (self-loop), agg1=0, out=0
__global__ void gcn_init(float* __restrict__ deg, float* __restrict__ agg1,
                         float* __restrict__ out, int nNodes) {
  int idx = blockIdx.x * blockDim.x + threadIdx.x;
  if (idx < nNodes * HID) agg1[idx] = 0.0f;
  if (idx < nNodes)       deg[idx]  = 1.0f;
  if (idx < nNodes * NCLS) out[idx] = 0.0f;
}

// degree count over target nodes
__global__ void gcn_degree(const int* __restrict__ col, float* __restrict__ deg, int E) {
  int e = blockIdx.x * blockDim.x + threadIdx.x;
  if (e < E) atomicAdd(&deg[col[e]], 1.0f);
}

// deg -> d^{-1/2} in place
__global__ void gcn_dinv(float* __restrict__ deg, int nNodes) {
  int i = blockIdx.x * blockDim.x + threadIdx.x;
  if (i < nNodes) {
    float d = deg[i];
    deg[i] = (d > 0.0f) ? rsqrtf(d) : 0.0f;
  }
}

// ---------------------------------------------------------------------------
// Pre-pack W1 [512x64] f32 into WMMA B-fragment order, f16:
// w1p[((kb*4 + tile)*32 + lane)*16 + i] = W1[kb*32 + (lane>=16?8:0) + (i&7) + (i>>3)*16][tile*16 + (lane&15)]
__global__ void gcn_pack_w1(const float* __restrict__ W1, _Float16* __restrict__ w1p) {
  int t = blockIdx.x * blockDim.x + threadIdx.x;
  if (t >= (F_IN / 32) * 4 * 32 * 16) return;
  int i    = t & 15;
  int lane = (t >> 4) & 31;
  int tile = (t >> 9) & 3;
  int kb   = t >> 11;
  int K = kb * 32 + ((lane >= 16) ? 8 : 0) + (i & 7) + ((i >> 3) << 4);
  int n = tile * 16 + (lane & 15);
  w1p[t] = (_Float16)W1[K * HID + n];
}

// Pre-pack W2 [64x16]: w2p[(kb*32 + lane)*16 + i]
__global__ void gcn_pack_w2(const float* __restrict__ W2, _Float16* __restrict__ w2p) {
  int t = blockIdx.x * blockDim.x + threadIdx.x;
  if (t >= 2 * 32 * 16) return;
  int i    = t & 15;
  int lane = (t >> 4) & 31;
  int kb   = t >> 9;
  int K = kb * 32 + ((lane >= 16) ? 8 : 0) + (i & 7) + ((i >> 3) << 4);
  int n = lane & 15;
  w2p[t] = (_Float16)W2[K * NCLS + n];
}

// ---------------------------------------------------------------------------
// GEMM1: h[N,64] = X[N,512] @ W1 ; 8 waves/block, one 16x64 panel per wave.
// Per k-step: A loads + all 4 B-fragment loads issue before any WMMA, so only
// the first WMMA pays a full wait; compiler + 8-wave occupancy hide the rest.
__global__ void gcn_gemm1(const float* __restrict__ X, const _Float16* __restrict__ w1p,
                          float* __restrict__ h, int nNodes) {
  const int lane = threadIdx.x & 31;
  const int wave = threadIdx.x >> 5;
  const int base = (blockIdx.x * 8 + wave) * 16;
  if (base >= nNodes) return;                 // wave-uniform exit; EXEC stays full

  const int mrow = lane & 15;
  int row = base + mrow;
  if (row >= nNodes) row = nNodes - 1;        // clamp loads, mask stores later
  const int koff = (lane < 16) ? 0 : 8;
  const float* xrow = X + (size_t)row * F_IN;
  const v16h* bv = (const v16h*)w1p;

  v8f acc[4] = {{}, {}, {}, {}};

#pragma unroll 4
  for (int kb = 0; kb < F_IN / 32; ++kb) {
    const int k0 = kb * 32;
    if (kb + 1 < F_IN / 32)
      __builtin_prefetch(xrow + k0 + 32 + koff, 0, 3);   // global_prefetch_b8 (near)

    // A-operand raw loads (4x b128) ...
    v4f p0 = *(const v4f*)(xrow + k0 + koff);
    v4f p1 = *(const v4f*)(xrow + k0 + koff + 4);
    v4f p2 = *(const v4f*)(xrow + k0 + 16 + koff);
    v4f p3 = *(const v4f*)(xrow + k0 + 16 + koff + 4);

    // ... then all 4 B fragments (8x b128) before any WMMA issues.
    v16h b[4];
#pragma unroll
    for (int t = 0; t < 4; ++t) b[t] = bv[(kb * 4 + t) * 32 + lane];

    v16h a;
#pragma unroll
    for (int i = 0; i < 4; ++i) {
      a[i]      = (_Float16)p0[i];
      a[4 + i]  = (_Float16)p1[i];
      a[8 + i]  = (_Float16)p2[i];
      a[12 + i] = (_Float16)p3[i];
    }

#pragma unroll
    for (int t = 0; t < 4; ++t) {
      acc[t] = __builtin_amdgcn_wmma_f32_16x16x32_f16(
          false, a, false, b[t], (short)0, acc[t], false, false);
    }
  }

  // C layout: VGPR r -> M = r (+8 for lanes 16..31), N = lane&15 within tile
  const int madd = (lane < 16) ? 0 : 8;
#pragma unroll
  for (int t = 0; t < 4; ++t) {
    const int n = t * 16 + mrow;
#pragma unroll
    for (int r = 0; r < 8; ++r) {
      int m = base + madd + r;
      if (m < nNodes) h[(size_t)m * HID + n] = acc[t][r];
    }
  }
}

// ---------------------------------------------------------------------------
// scatter1: agg1[col] += h[row] * norm   (16 threads x 4 floats per edge;
// index/dinv loads are same-address broadcasts within each 16-lane group)
__global__ void gcn_scatter1(const int* __restrict__ rowi, const int* __restrict__ coli,
                             const float* __restrict__ dinv, const float* __restrict__ h,
                             float* __restrict__ agg, int E) {
  long long t = (long long)blockIdx.x * blockDim.x + threadIdx.x;
  if (t >= (long long)E * 16) return;
  int e  = (int)(t >> 4);
  int j0 = ((int)t & 15) * 4;
  int r = rowi[e];
  int c = coli[e];
  float nrm = dinv[r] * dinv[c];
  v4f hv = *(const v4f*)(h + (size_t)r * HID + j0);
  float* dst = agg + (size_t)c * HID + j0;
  atomicAdd(dst + 0, hv[0] * nrm);
  atomicAdd(dst + 1, hv[1] * nrm);
  atomicAdd(dst + 2, hv[2] * nrm);
  atomicAdd(dst + 3, hv[3] * nrm);
}

// relu( agg1 + self-loop + b1 ) -> xh (f16, ready as GEMM2 A operand)
__global__ void gcn_relu_bias(const float* __restrict__ agg, const float* __restrict__ h,
                              const float* __restrict__ dinv, const float* __restrict__ b1,
                              _Float16* __restrict__ xh, int nNodes) {
  int idx = blockIdx.x * blockDim.x + threadIdx.x;
  if (idx >= nNodes * HID) return;
  int n = idx >> 6;
  int j = idx & (HID - 1);
  float dv = dinv[n];
  float v = agg[idx] + h[idx] * dv * dv + b1[j];
  xh[idx] = (_Float16)fmaxf(v, 0.0f);
}

// ---------------------------------------------------------------------------
// GEMM2: h2[N,16] = x[N,64] @ W2 ; one 16x16 tile per wave, K=64 fully
// unrolled with both B fragments preloaded (single wait covers both WMMAs).
__global__ void gcn_gemm2(const _Float16* __restrict__ xh, const _Float16* __restrict__ w2p,
                          float* __restrict__ h2, int nNodes) {
  const int lane = threadIdx.x & 31;
  const int wave = threadIdx.x >> 5;
  const int base = (blockIdx.x * 8 + wave) * 16;
  if (base >= nNodes) return;

  const int mrow = lane & 15;
  int row = base + mrow;
  if (row >= nNodes) row = nNodes - 1;
  const int koff = (lane < 16) ? 0 : 8;
  const _Float16* xr = xh + (size_t)row * HID;
  const v16h* bv = (const v16h*)w2p;

  v16h b0 = bv[lane];
  v16h b1 = bv[32 + lane];
  v8h q00 = *(const v8h*)(xr + koff);
  v8h q01 = *(const v8h*)(xr + 16 + koff);
  v8h q10 = *(const v8h*)(xr + 32 + koff);
  v8h q11 = *(const v8h*)(xr + 48 + koff);

  v16h a0, a1;
#pragma unroll
  for (int i = 0; i < 8; ++i) {
    a0[i] = q00[i]; a0[8 + i] = q01[i];
    a1[i] = q10[i]; a1[8 + i] = q11[i];
  }

  v8f acc = {};
  acc = __builtin_amdgcn_wmma_f32_16x16x32_f16(false, a0, false, b0, (short)0, acc, false, false);
  acc = __builtin_amdgcn_wmma_f32_16x16x32_f16(false, a1, false, b1, (short)0, acc, false, false);

  const int madd = (lane < 16) ? 0 : 8;
#pragma unroll
  for (int r = 0; r < 8; ++r) {
    int m = base + madd + r;
    if (m < nNodes) h2[(size_t)m * NCLS + mrow] = acc[r];
  }
}

// scatter2: out[col] += h2[row] * norm   (4 threads x 4 floats per edge)
__global__ void gcn_scatter2(const int* __restrict__ rowi, const int* __restrict__ coli,
                             const float* __restrict__ dinv, const float* __restrict__ h2,
                             float* __restrict__ out, int E) {
  long long t = (long long)blockIdx.x * blockDim.x + threadIdx.x;
  if (t >= (long long)E * 4) return;
  int e  = (int)(t >> 2);
  int j0 = ((int)t & 3) * 4;
  int r = rowi[e];
  int c = coli[e];
  float nrm = dinv[r] * dinv[c];
  v4f hv = *(const v4f*)(h2 + (size_t)r * NCLS + j0);
  float* dst = out + (size_t)c * NCLS + j0;
  atomicAdd(dst + 0, hv[0] * nrm);
  atomicAdd(dst + 1, hv[1] * nrm);
  atomicAdd(dst + 2, hv[2] * nrm);
  atomicAdd(dst + 3, hv[3] * nrm);
}

// out += self-loop contribution + b2
__global__ void gcn_final(const float* __restrict__ h2, const float* __restrict__ dinv,
                          const float* __restrict__ b2, float* __restrict__ out, int nNodes) {
  int idx = blockIdx.x * blockDim.x + threadIdx.x;
  if (idx >= nNodes * NCLS) return;
  int n = idx >> 4;
  int j = idx & (NCLS - 1);
  float dv = dinv[n];
  out[idx] += h2[idx] * dv * dv + b2[j];
}

// ---------------------------------------------------------------------------
extern "C" void kernel_launch(void* const* d_in, const int* in_sizes, int n_in,
                              void* d_out, int out_size, void* d_ws, size_t ws_size,
                              hipStream_t stream) {
  const float* X   = (const float*)d_in[0];
  const int*   ei  = (const int*)d_in[1];
  const float* W1  = (const float*)d_in[2];
  const float* b1  = (const float*)d_in[3];
  const float* W2  = (const float*)d_in[4];
  const float* b2  = (const float*)d_in[5];
  float* out = (float*)d_out;

  const int nNodes = in_sizes[0] / F_IN;   // 100000
  const int E      = in_sizes[1] / 2;      // 1600000
  const int* rowi = ei;
  const int* coli = ei + E;

  // Workspace carve-out (256B-aligned regions)
  char* p = (char*)d_ws;
  auto carve = [&](size_t bytes) -> char* {
    char* r = p;
    p += (bytes + 255) & ~(size_t)255;
    return r;
  };
  float*     deg  = (float*)carve((size_t)nNodes * 4);            // deg -> dinv
  float*     h    = (float*)carve((size_t)nNodes * HID * 4);      // X@W1 ; reused as h2
  float*     agg1 = (float*)carve((size_t)nNodes * HID * 4);
  _Float16*  xh   = (_Float16*)carve((size_t)nNodes * HID * 2);
  _Float16*  w1p  = (_Float16*)carve((size_t)(F_IN / 32) * 4 * 32 * 16 * 2);
  _Float16*  w2p  = (_Float16*)carve((size_t)2 * 32 * 16 * 2);
  float*     h2   = h;   // h is dead once xh is built; reuse for layer-2 GEMM out

  const int T = 256;
  const int nPanels = (nNodes + 127) / 128;   // 8 waves x 16 rows per block

  gcn_init<<<(nNodes * HID + T - 1) / T, T, 0, stream>>>(deg, agg1, out, nNodes);
  gcn_pack_w1<<<((F_IN / 32) * 4 * 32 * 16 + T - 1) / T, T, 0, stream>>>(W1, w1p);
  gcn_pack_w2<<<(2 * 32 * 16 + T - 1) / T, T, 0, stream>>>(W2, w2p);
  gcn_degree<<<(E + T - 1) / T, T, 0, stream>>>(coli, deg, E);
  gcn_dinv<<<(nNodes + T - 1) / T, T, 0, stream>>>(deg, nNodes);

  gcn_gemm1<<<nPanels, T, 0, stream>>>(X, w1p, h, nNodes);
  {
    long long tot = (long long)E * 16;
    gcn_scatter1<<<(unsigned)((tot + T - 1) / T), T, 0, stream>>>(rowi, coli, deg, h, agg1, E);
  }
  gcn_relu_bias<<<(nNodes * HID + T - 1) / T, T, 0, stream>>>(agg1, h, deg, b1, xh, nNodes);

  gcn_gemm2<<<nPanels, T, 0, stream>>>(xh, w2p, h2, nNodes);
  {
    long long tot = (long long)E * 4;
    gcn_scatter2<<<(unsigned)((tot + T - 1) / T), T, 0, stream>>>(rowi, coli, deg, h2, out, E);
  }
  gcn_final<<<(nNodes * NCLS + T - 1) / T, T, 0, stream>>>(h2, deg, b2, out, nNodes);
}
